// BiLSTMSeq2Seq_37512244363825
// MI455X (gfx1250) — matrix operand order repaired and checked by reading, not compiled, not run on hardware
//
#include <hip/hip_runtime.h>
#include <hip/hip_bf16.h>
#include <cstdint>

// ---------------------------------------------------------------------------
// Problem dims (compile-time)
// ---------------------------------------------------------------------------
#define BD 32      // batch
#define SD 128     // encoder seq
#define TD 64      // decoder seq
#define ED 256     // embed dim
#define HD 512     // hidden dim
#define GD 2048    // 4*H gate dim
#define EVD 32000  // output vocab

typedef __attribute__((ext_vector_type(16))) __bf16 v16bf;
typedef __attribute__((ext_vector_type(8)))  float  v8f;

// ---------------------------------------------------------------------------
// WMMA fragment loads, laid out per CDNA5 ISA 7.12.2 (wave32)
// A (16x32 bf16, MxK): lane<16 -> M=m0+lane, K = [k0..k0+7] U [k0+16..k0+23]
//                      lane>=16 -> M=m0+lane-16, K = [k0+8..] U [k0+24..]
// ---------------------------------------------------------------------------
__device__ __forceinline__ v16bf load_frag_a(const __bf16* __restrict__ base,
                                             int ld, int m0, int k0, int lane) {
  int row  = m0 + (lane & 15);
  int koff = k0 + ((lane >> 4) << 3);
  const __bf16* p = base + (size_t)row * ld + koff;
  union { v16bf v; uint4 q[2]; } u;
  u.q[0] = *reinterpret_cast<const uint4*>(p);
  u.q[1] = *reinterpret_cast<const uint4*>(p + 16);
  return u.v;
}

// B (32x16 bf16, KxN) from row-major W[N][K]: B[k][j] = W[n0+j][k0+k]
// lane<16 -> N=n0+lane, K=[k0..k0+7] U [k0+16..k0+23]; lane>=16 -> N=n0+lane-16, K=[k0+8..] U [k0+24..]
__device__ __forceinline__ v16bf load_frag_b(const __bf16* __restrict__ base,
                                             int ld, int n0, int k0, int lane) {
  int row  = n0 + (lane & 15);
  int koff = k0 + ((lane >> 4) << 3);
  const __bf16* p = base + (size_t)row * ld + koff;
  union { v16bf v; uint4 q[2]; } u;
  u.q[0] = *reinterpret_cast<const uint4*>(p);
  u.q[1] = *reinterpret_cast<const uint4*>(p + 16);
  return u.v;
}

__device__ __forceinline__ float sigm_f(float x) {
  return 1.0f / (1.0f + __expf(-x));
}
__device__ __forceinline__ float tanh_f(float x) {
  return 2.0f / (1.0f + __expf(-2.0f * x)) - 1.0f;
}

// ---------------------------------------------------------------------------
// Small utility kernels
// ---------------------------------------------------------------------------
__global__ void f32_to_bf16_kernel(const float* __restrict__ in,
                                   __bf16* __restrict__ out, int n) {
  int i = blockIdx.x * blockDim.x + threadIdx.x;
  if (i < n) out[i] = (__bf16)in[i];
}

// out[s*batch*edim + b*edim + e] = emb[idx[b*seq + s]][e]  (as bf16)
__global__ void gather_emb_bf16(const float* __restrict__ emb,
                                const int* __restrict__ idx,
                                __bf16* __restrict__ out,
                                int seq, int batch, int edim) {
  int i = blockIdx.x * blockDim.x + threadIdx.x;
  int total = seq * batch * edim;
  if (i >= total) return;
  int e  = i % edim;
  int sb = i / edim;
  int b  = sb % batch;
  int s  = sb / batch;
  out[i] = (__bf16)emb[(size_t)idx[b * seq + s] * edim + e];
}

__global__ void init_state_kernel(const float* __restrict__ h0,
                                  const float* __restrict__ c0,
                                  __bf16* __restrict__ ha, __bf16* __restrict__ hb,
                                  float* __restrict__ ca, float* __restrict__ cb,
                                  int n) {
  int i = blockIdx.x * blockDim.x + threadIdx.x;
  if (i >= n) return;
  __bf16 hv = (__bf16)h0[i];
  ha[i] = hv; hb[i] = hv;
  float cv = c0[i];
  ca[i] = cv; cb[i] = cv;
}

// ---------------------------------------------------------------------------
// Fused LSTM cell: one block = 16 hidden columns, 4 waves = 4 gate tiles
// gates[32,16] per gate accumulated via v_wmma_f32_16x16x32_bf16 into LDS,
// then the same block does the elementwise cell update.
// Launch: <<<32, 128>>>
// ---------------------------------------------------------------------------
__global__ __launch_bounds__(128)
void lstm_cell_wmma(const __bf16* __restrict__ x, int kx,
                    const __bf16* __restrict__ hin,
                    const __bf16* __restrict__ Wih,   // [GD, kx] bf16
                    const __bf16* __restrict__ Whh,   // [GD, HD] bf16
                    const float*  __restrict__ bih,
                    const float*  __restrict__ bhh,
                    float*  __restrict__ c,           // [BD, HD] in/out
                    __bf16* __restrict__ hout,        // [BD, HD]
                    __bf16* __restrict__ hs_save) {   // optional [BD, HD]
  __shared__ float sg[4][BD][16];

  int wave = threadIdx.x >> 5;
  int lane = threadIdx.x & 31;
  int col0 = blockIdx.x * 16;        // hidden columns [col0, col0+16)
  int n0   = wave * HD + col0;       // gate-tile base row in W (i/f/g/o)

  v8f acc0 = {};  // M rows 0..15
  v8f acc1 = {};  // M rows 16..31

  // x @ Wih^T
  for (int k0 = 0; k0 < kx; k0 += 32) {
    v16bf a0 = load_frag_a(x, kx, 0, k0, lane);
    v16bf a1 = load_frag_a(x, kx, 16, k0, lane);
    v16bf bm = load_frag_b(Wih, kx, n0, k0, lane);
    acc0 = __builtin_amdgcn_wmma_f32_16x16x32_bf16(false, a0, false, bm,
                                                   (short)0, acc0, false, false);
    acc1 = __builtin_amdgcn_wmma_f32_16x16x32_bf16(false, a1, false, bm,
                                                   (short)0, acc1, false, false);
  }
  // h @ Whh^T
  for (int k0 = 0; k0 < HD; k0 += 32) {
    v16bf a0 = load_frag_a(hin, HD, 0, k0, lane);
    v16bf a1 = load_frag_a(hin, HD, 16, k0, lane);
    v16bf bm = load_frag_b(Whh, HD, n0, k0, lane);
    acc0 = __builtin_amdgcn_wmma_f32_16x16x32_bf16(false, a0, false, bm,
                                                   (short)0, acc0, false, false);
    acc1 = __builtin_amdgcn_wmma_f32_16x16x32_bf16(false, a1, false, bm,
                                                   (short)0, acc1, false, false);
  }

  // bias + stash gates in LDS (C/D layout: lane<16 col=lane rows m0+0..7,
  // lane>=16 col=lane-16 rows m0+8..15)
  int col  = lane & 15;
  float bias = bih[n0 + col] + bhh[n0 + col];
  int rb = (lane >> 4) << 3;
#pragma unroll
  for (int v = 0; v < 8; v++) {
    sg[wave][rb + v][col]      = acc0[v] + bias;
    sg[wave][16 + rb + v][col] = acc1[v] + bias;
  }
  __syncthreads();

  // elementwise cell update: 128 threads x 4 batches each
  int pc = threadIdx.x & 15;
  int bq = (threadIdx.x >> 4) * 4;
#pragma unroll
  for (int bb = 0; bb < 4; bb++) {
    int b = bq + bb;
    float ig = sg[0][b][pc];
    float fg = sg[1][b][pc];
    float gg = sg[2][b][pc];
    float og = sg[3][b][pc];
    int idx = b * HD + col0 + pc;
    float cn = sigm_f(fg) * c[idx] + sigm_f(ig) * tanh_f(gg);
    float hn = sigm_f(og) * tanh_f(cn);
    c[idx] = cn;
    __bf16 hb = (__bf16)hn;
    hout[idx] = hb;
    if (hs_save) hs_save[idx] = hb;
  }
}

// ---------------------------------------------------------------------------
// Projection: logits[r, v] = relu(hs[r,:] @ Wout[v,:] + bout[v])
// hs rows r = t*32 + b; output index = b*T*EV + t*EV + v (f32)
// Launch: <<<dim3(EVD/128, 2048/16), 256>>>  (8 waves, 16 N-cols each)
// ---------------------------------------------------------------------------
__global__ __launch_bounds__(256)
void proj_wmma(const __bf16* __restrict__ hs,     // [T*B, HD] bf16
               const __bf16* __restrict__ Wout,   // [EVD, HD] bf16
               const float*  __restrict__ bout,
               float* __restrict__ out) {
  int wave = threadIdx.x >> 5;
  int lane = threadIdx.x & 31;
  int v0 = blockIdx.x * 128 + wave * 16;
  int r0 = blockIdx.y * 16;

  v8f acc = {};
  for (int k0 = 0; k0 < HD; k0 += 32) {
    v16bf a = load_frag_a(hs, HD, r0, k0, lane);
    v16bf b = load_frag_b(Wout, HD, v0, k0, lane);
    if (k0 + 32 < HD)  // -> global_prefetch_b8 for next weight chunk
      __builtin_prefetch(Wout + (size_t)(v0 + (lane & 15)) * HD + k0 + 32, 0, 1);
    acc = __builtin_amdgcn_wmma_f32_16x16x32_bf16(false, a, false, b,
                                                  (short)0, acc, false, false);
  }

  int col = v0 + (lane & 15);
  float bb = bout[col];
  int rbase = r0 + ((lane >> 4) << 3);
#pragma unroll
  for (int v = 0; v < 8; v++) {
    int r = rbase + v;
    int t = r >> 5;        // r = t*32 + b
    int b = r & 31;
    out[(size_t)b * TD * EVD + (size_t)t * EVD + col] = fmaxf(acc[v] + bb, 0.0f);
  }
}

// ---------------------------------------------------------------------------
// Per-(b,t) row stats over V for the axis=2 log_softmax.
// One block of 256 per row; 2048 rows.
// ---------------------------------------------------------------------------
__global__ __launch_bounds__(256)
void row_stats(const float* __restrict__ out,
               float* __restrict__ rmax, float* __restrict__ rlse) {
  __shared__ float red[256];
  int r = blockIdx.x;
  const float* row = out + (size_t)r * EVD;

  float m = -3.4e38f;
  for (int v = threadIdx.x; v < EVD; v += 256) m = fmaxf(m, row[v]);
  red[threadIdx.x] = m;
  __syncthreads();
  for (int s = 128; s > 0; s >>= 1) {
    if (threadIdx.x < s)
      red[threadIdx.x] = fmaxf(red[threadIdx.x], red[threadIdx.x + s]);
    __syncthreads();
  }
  m = red[0];
  __syncthreads();

  float ssum = 0.0f;
  for (int v = threadIdx.x; v < EVD; v += 256) ssum += __expf(row[v] - m);
  red[threadIdx.x] = ssum;
  __syncthreads();
  for (int s = 128; s > 0; s >>= 1) {
    if (threadIdx.x < s) red[threadIdx.x] += red[threadIdx.x + s];
    __syncthreads();
  }
  if (threadIdx.x == 0) { rmax[r] = m; rlse[r] = __logf(red[0]); }
}

// ---------------------------------------------------------------------------
// Fused: apply axis=2 normalization, then log_softmax over axis=0 (batch).
// One thread per (t, v); reads/writes its own 32 batch entries (in-place ok).
// ---------------------------------------------------------------------------
__global__ __launch_bounds__(256)
void final_lsm(float* __restrict__ out,
               const float* __restrict__ rmax, const float* __restrict__ rlse) {
  size_t gid = (size_t)blockIdx.x * blockDim.x + threadIdx.x;
  if (gid >= (size_t)TD * EVD) return;
  int t = (int)(gid / EVD);
  int v = (int)(gid % EVD);

  float y[BD];
  float m2 = -3.4e38f;
#pragma unroll 4
  for (int b = 0; b < BD; b++) {
    size_t idx = (size_t)b * TD * EVD + (size_t)t * EVD + v;
    int r = b * TD + t;
    float yy = out[idx] - rmax[r] - rlse[r];
    y[b] = yy;
    m2 = fmaxf(m2, yy);
  }
  float s2 = 0.0f;
#pragma unroll 4
  for (int b = 0; b < BD; b++) s2 += __expf(y[b] - m2);
  float l2 = __logf(s2);
#pragma unroll 4
  for (int b = 0; b < BD; b++) {
    size_t idx = (size_t)b * TD * EVD + (size_t)t * EVD + v;
    out[idx] = y[b] - m2 - l2;
  }
}

// ---------------------------------------------------------------------------
// Host orchestration
// ---------------------------------------------------------------------------
extern "C" void kernel_launch(void* const* d_in, const int* in_sizes, int n_in,
                              void* d_out, int out_size, void* d_ws, size_t ws_size,
                              hipStream_t stream) {
  (void)in_sizes; (void)n_in; (void)out_size; (void)ws_size;

  const int*   inp     = (const int*)d_in[0];
  const int*   tar     = (const int*)d_in[1];
  const float* enc_emb = (const float*)d_in[2];
  const float* dec_emb = (const float*)d_in[3];
  const float* Wih_fw  = (const float*)d_in[4];
  const float* Whh_fw  = (const float*)d_in[5];
  const float* bih_fw  = (const float*)d_in[6];
  const float* bhh_fw  = (const float*)d_in[7];
  const float* Wih_bw  = (const float*)d_in[8];
  const float* Whh_bw  = (const float*)d_in[9];
  const float* bih_bw  = (const float*)d_in[10];
  const float* bhh_bw  = (const float*)d_in[11];
  const float* Wih_d1  = (const float*)d_in[12];
  const float* Whh_d1  = (const float*)d_in[13];
  const float* bih_d1  = (const float*)d_in[14];
  const float* bhh_d1  = (const float*)d_in[15];
  const float* Wih_d2  = (const float*)d_in[16];
  const float* Whh_d2  = (const float*)d_in[17];
  const float* bih_d2  = (const float*)d_in[18];
  const float* bhh_d2  = (const float*)d_in[19];
  const float* Wout    = (const float*)d_in[20];
  const float* bout    = (const float*)d_in[21];
  const float* init_h  = (const float*)d_in[22];
  const float* init_c  = (const float*)d_in[23];
  float* out = (float*)d_out;

  // bump allocator over workspace
  char* base = (char*)d_ws;
  size_t off = 0;
  auto alloc = [&](size_t bytes) -> void* {
    void* p = base + off;
    off += (bytes + 255) & ~(size_t)255;
    return p;
  };

  __bf16* wih_fw_b = (__bf16*)alloc((size_t)GD * ED * 2);
  __bf16* whh_fw_b = (__bf16*)alloc((size_t)GD * HD * 2);
  __bf16* wih_bw_b = (__bf16*)alloc((size_t)GD * HD * 2);
  __bf16* whh_bw_b = (__bf16*)alloc((size_t)GD * HD * 2);
  __bf16* wih_d1_b = (__bf16*)alloc((size_t)GD * ED * 2);
  __bf16* whh_d1_b = (__bf16*)alloc((size_t)GD * HD * 2);
  __bf16* wih_d2_b = (__bf16*)alloc((size_t)GD * HD * 2);
  __bf16* whh_d2_b = (__bf16*)alloc((size_t)GD * HD * 2);
  __bf16* wout_b   = (__bf16*)alloc((size_t)EVD * HD * 2);
  __bf16* xenc     = (__bf16*)alloc((size_t)SD * BD * ED * 2);
  __bf16* xdec     = (__bf16*)alloc((size_t)TD * BD * ED * 2);
  __bf16* hfw[2]   = { (__bf16*)alloc((size_t)BD * HD * 2),
                       (__bf16*)alloc((size_t)BD * HD * 2) };
  __bf16* hbw[2]   = { (__bf16*)alloc((size_t)BD * HD * 2),
                       (__bf16*)alloc((size_t)BD * HD * 2) };
  float*  cA       = (float*)alloc((size_t)BD * HD * 4);
  float*  cB       = (float*)alloc((size_t)BD * HD * 4);
  __bf16* hsbuf    = (__bf16*)alloc((size_t)TD * BD * HD * 2);
  float*  rmax     = (float*)alloc((size_t)BD * TD * 4);
  float*  rlse     = (float*)alloc((size_t)BD * TD * 4);

  // --- 1. weights -> bf16 ---
  struct { const float* s; __bf16* d; int n; } cv[9] = {
    { Wih_fw, wih_fw_b, GD * ED }, { Whh_fw, whh_fw_b, GD * HD },
    { Wih_bw, wih_bw_b, GD * HD }, { Whh_bw, whh_bw_b, GD * HD },
    { Wih_d1, wih_d1_b, GD * ED }, { Whh_d1, whh_d1_b, GD * HD },
    { Wih_d2, wih_d2_b, GD * HD }, { Whh_d2, whh_d2_b, GD * HD },
    { Wout,   wout_b,   EVD * HD },
  };
  for (int i = 0; i < 9; i++)
    f32_to_bf16_kernel<<<(cv[i].n + 255) / 256, 256, 0, stream>>>(cv[i].s, cv[i].d, cv[i].n);

  // --- 2. embedding gathers (-> [seq, B, E] bf16) ---
  gather_emb_bf16<<<(SD * BD * ED + 255) / 256, 256, 0, stream>>>(
      enc_emb, inp, xenc, SD, BD, ED);
  gather_emb_bf16<<<(TD * BD * ED + 255) / 256, 256, 0, stream>>>(
      dec_emb, tar, xdec, TD, BD, ED);

  // --- 3. init states ---
  init_state_kernel<<<(BD * HD + 255) / 256, 256, 0, stream>>>(
      init_h, init_c, hfw[0], hbw[0], cA, cB, BD * HD);

  // --- 4. encoder forward: x_t from embeddings ---
  int p = 0;
  for (int s = 0; s < SD; s++) {
    lstm_cell_wmma<<<HD / 16, 128, 0, stream>>>(
        xenc + (size_t)s * BD * ED, ED, hfw[p],
        wih_fw_b, whh_fw_b, bih_fw, bhh_fw,
        cA, hfw[p ^ 1], (  __bf16*)nullptr);
    p ^= 1;
  }
  // after SD (even) steps, final fw hidden is in hfw[0]

  // --- 5. encoder backward: h feeds itself as input ---
  int q = 0;
  for (int s = 0; s < SD; s++) {
    lstm_cell_wmma<<<HD / 16, 128, 0, stream>>>(
        hbw[q], HD, hbw[q],
        wih_bw_b, whh_bw_b, bih_bw, bhh_bw,
        cB, hbw[q ^ 1], (__bf16*)nullptr);
    q ^= 1;
  }
  // final bw cell state is in cB

  // --- 6. decoder: init (h_fw, c_bw); two stacked cells per step ---
  __bf16* htmp = hbw[0];   // safe to clobber now
  p = 0;                   // decoder h lives in hfw[p]
  for (int t = 0; t < TD; t++) {
    lstm_cell_wmma<<<HD / 16, 128, 0, stream>>>(
        xdec + (size_t)t * BD * ED, ED, hfw[p],
        wih_d1_b, whh_d1_b, bih_d1, bhh_d1,
        cB, htmp, (__bf16*)nullptr);
    lstm_cell_wmma<<<HD / 16, 128, 0, stream>>>(
        htmp, HD, htmp,
        wih_d2_b, whh_d2_b, bih_d2, bhh_d2,
        cB, hfw[p ^ 1], hsbuf + (size_t)t * BD * HD);
    p ^= 1;
  }

  // --- 7. output projection (WMMA) + bias + relu -> d_out ---
  proj_wmma<<<dim3(EVD / 128, (TD * BD) / 16), 256, 0, stream>>>(
      hsbuf, wout_b, bout, out);

  // --- 8. axis=2 log_softmax stats per (b,t) row ---
  row_stats<<<BD * TD, 256, 0, stream>>>(out, rmax, rlse);

  // --- 9. fused axis=2 normalize + axis=0 log_softmax ---
  final_lsm<<<((size_t)TD * EVD + 255) / 256, 256, 0, stream>>>(out, rmax, rlse);
}
